// MambaBlock_2680059593337
// MI455X (gfx1250) — compile-verified, matrix operand order checked by reading, not run on hardware
//
#include <hip/hip_runtime.h>
#include <hip/hip_bf16.h>
#include <math.h>

// Problem constants (from reference)
#define LSEQ    2048
#define DMODEL  1024
#define DIN     2048
#define NSTATE  16
#define DTRANK  64
#define KCONV   4
#define SCAN_CHUNK 8   // timesteps of B/C staged per LDS barrier pair

typedef __attribute__((ext_vector_type(16))) _Float16 v16h;
typedef __attribute__((ext_vector_type(8)))  _Float16 v8h;
typedef __attribute__((ext_vector_type(4)))  _Float16 v4h;
typedef __attribute__((ext_vector_type(8)))  float    v8f;
typedef __attribute__((ext_vector_type(4)))  float    v4f;

// Global (address_space(1)) pointer types: forces global_load_b128 (single
// LOADcnt) instead of flat_load_b128 (LOADcnt+DScnt, ties up LDS path) when
// fragment pointers are carried through arrays.
typedef __attribute__((address_space(1))) const _Float16 GF16;
typedef __attribute__((address_space(1))) const v8h      GV8H;

union FragU { v16h v; v8h h[2]; };

__device__ __forceinline__ float swish_f(float x) {
  return x / (1.0f + __expf(-x));
}
__device__ __forceinline__ float softplus_f(float x) {
  return (x > 20.0f) ? x : log1pf(__expf(x));
}

// ---------------------------------------------------------------------------
// Generic WMMA GEMM: C(MxN,f32) = A(MxK,f16,row-major) * Bt(NxK,f16)^T
// One wave per block; wave computes (MT*16) x (NT*16) in registers.
// Per-lane global fragment pointers are computed once and bumped 64B/K-step.
// ISA 16-bit A 16x32 layout: lane = m + 16*kg; elems 0..7 -> k0+kg*8,
// elems 8..15 -> k0+16+kg*8. B fragments use the same layout on Bt (NxK).
// MODE: 0 = plain, 1 = +bias, 2 = softplus(x + bias)
// ---------------------------------------------------------------------------
template<int MT, int NT, int MODE>
__global__ void gemm_wmma_f16(const _Float16* __restrict__ A,
                              const _Float16* __restrict__ Bt,
                              float* __restrict__ C,
                              const float* __restrict__ bias,
                              int M, int N, int K) {
  const int col0 = blockIdx.x * (NT * 16);
  const int row0 = blockIdx.y * (MT * 16);
  const int lane = threadIdx.x & 31;
  const int m    = lane & 15;
  const int kg   = lane >> 4;   // 0 or 1

  GF16* ap[MT];
  GF16* bp[NT];
#pragma unroll
  for (int i = 0; i < MT; ++i)
    ap[i] = (GF16*)A + (size_t)(row0 + i * 16 + m) * K + kg * 8;
#pragma unroll
  for (int j = 0; j < NT; ++j)
    bp[j] = (GF16*)Bt + (size_t)(col0 + j * 16 + m) * K + kg * 8;

  const v8f zero = {0.f, 0.f, 0.f, 0.f, 0.f, 0.f, 0.f, 0.f};
  v8f acc[MT][NT];
#pragma unroll
  for (int i = 0; i < MT; ++i)
#pragma unroll
    for (int j = 0; j < NT; ++j) acc[i][j] = zero;

  for (int k0 = 0; k0 < K; k0 += 32) {
    v16h a[MT], b[NT];
#pragma unroll
    for (int i = 0; i < MT; ++i) {
      FragU f;
      f.h[0] = *(GV8H*)(ap[i]);
      f.h[1] = *(GV8H*)(ap[i] + 16);
      a[i] = f.v;
      if (i == 0)  // one speculative prefetch per matrix per K-step
        __builtin_prefetch((const void*)(const _Float16*)(ap[0] + 64), 0, 1);
      ap[i] += 32;
    }
#pragma unroll
    for (int j = 0; j < NT; ++j) {
      FragU f;
      f.h[0] = *(GV8H*)(bp[j]);
      f.h[1] = *(GV8H*)(bp[j] + 16);
      b[j] = f.v;
      if (j == 0)
        __builtin_prefetch((const void*)(const _Float16*)(bp[0] + 64), 0, 1);
      bp[j] += 32;
    }
#pragma unroll
    for (int i = 0; i < MT; ++i)
#pragma unroll
      for (int j = 0; j < NT; ++j)
        acc[i][j] = __builtin_amdgcn_wmma_f32_16x16x32_f16(
            false, a[i], false, b[j], (short)0, acc[i][j], false, false);
  }

  // C/D layout: VGPR r, lanes 0-15 -> M=r, N=lane; lanes 16-31 -> M=8+r.
#pragma unroll
  for (int i = 0; i < MT; ++i) {
#pragma unroll
    for (int j = 0; j < NT; ++j) {
      const int r0 = row0 + i * 16 + ((lane >> 4) << 3);
      const int c  = col0 + j * 16 + (lane & 15);
      float* cp = C + (size_t)r0 * N + c;
      const float bv = (MODE >= 1) ? bias[c] : 0.0f;
#pragma unroll
      for (int r = 0; r < 8; ++r) {
        float v = acc[i][j][r];
        if (MODE >= 1) v += bv;
        if (MODE == 2) v = softplus_f(v);
        cp[(size_t)r * N] = v;
      }
    }
  }
}

// ---------------------------------------------------------------------------
// Elementwise helpers
// ---------------------------------------------------------------------------

// 4-wide f32 -> f16 cast (n must be a multiple of 4; all our sizes are)
__global__ void cast_f32_f16_v4(const float* __restrict__ src,
                                _Float16* __restrict__ dst, int n4) {
  int i = blockIdx.x * blockDim.x + threadIdx.x;
  if (i < n4) {
    v4f s = ((const v4f*)src)[i];
    v4h d;
#pragma unroll
    for (int k = 0; k < 4; ++k) d[k] = (_Float16)s[k];
    ((v4h*)dst)[i] = d;
  }
}

// dst[c*R + r] = (f16) src[r*C + c]   (transpose + downconvert a weight)
__global__ void transpose_cast_f16(const float* __restrict__ src,
                                   _Float16* __restrict__ dst, int R, int C) {
  int i = blockIdx.x * blockDim.x + threadIdx.x;
  if (i < R * C) {
    int r = i / C, c = i - r * C;
    dst[(size_t)c * R + r] = (_Float16)src[(size_t)r * C + c];
  }
}

// dst (rows x cols f16, packed) = src[:, :cols] with leading dim src_ld.
// Vectorized 4-wide along cols (cols % 4 == 0, src_ld % 4 == 0).
__global__ void slice_cast_f16_v4(const float* __restrict__ src, int src_ld,
                                  _Float16* __restrict__ dst, int rows, int cols) {
  int i = blockIdx.x * blockDim.x + threadIdx.x;
  const int c4 = cols >> 2;
  if (i < rows * c4) {
    int r = i / c4, g = i - r * c4;
    v4f s = *(const v4f*)(src + (size_t)r * src_ld + g * 4);
    v4h d;
#pragma unroll
    for (int k = 0; k < 4; ++k) d[k] = (_Float16)s[k];
    ((v4h*)dst)[i] = d;
  }
}

// Depthwise causal conv1d (K=4) over xi = C1[:, 0:DIN] (ld 2*DIN), + bias,
// swish, write f32 (for scan) and f16 (for the x_dbl GEMM).
__global__ void conv_swish_kernel(const float* __restrict__ C1,
                                  const float* __restrict__ conv_w,
                                  const float* __restrict__ conv_b,
                                  float* __restrict__ xi_act,
                                  _Float16* __restrict__ xi_h) {
  int i = blockIdx.x * blockDim.x + threadIdx.x;   // l * DIN + d
  if (i >= LSEQ * DIN) return;
  int l = i / DIN, d = i - l * DIN;
  float acc = conv_b[d];
#pragma unroll
  for (int k = 0; k < KCONV; ++k) {
    int ls = l - (KCONV - 1) + k;
    if (ls >= 0)
      acc += conv_w[d * KCONV + k] * C1[(size_t)ls * (2 * DIN) + d];
  }
  float s = swish_f(acc);
  xi_act[i] = s;
  xi_h[i]   = (_Float16)s;
}

// ---------------------------------------------------------------------------
// Selective scan: one thread per channel d (16 states in VGPRs), serial in l.
// h_n <- exp(dt*A_n)*h_n + dt*u*B_n ; y = sum_n h_n*C_n + u*D ; y *= swish(res)
// B/C rows staged in LDS, SCAN_CHUNK timesteps per barrier pair
// (256 threads = SCAN_CHUNK x 32 elements, one element per thread).
// ---------------------------------------------------------------------------
__global__ void scan_kernel(const float* __restrict__ xi_act,   // L x DIN
                            const float* __restrict__ delta,    // L x DIN (softplus'd)
                            const float* __restrict__ x_dbl,    // L x 96
                            const float* __restrict__ res,      // C1 + DIN, ld 2*DIN
                            const float* __restrict__ A_log,    // DIN x N
                            const float* __restrict__ Dp,       // DIN
                            _Float16* __restrict__ y_h) {       // L x DIN
  const int d = blockIdx.x * blockDim.x + threadIdx.x;  // exactly DIN threads
  __shared__ float sBC[SCAN_CHUNK][2 * NSTATE];

  float Arow[NSTATE], h[NSTATE];
#pragma unroll
  for (int n = 0; n < NSTATE; ++n) {
    Arow[n] = -__expf(A_log[(size_t)d * NSTATE + n]);
    h[n] = 0.0f;
  }
  const float Dd = Dp[d];

  const int st = threadIdx.x >> 5;   // timestep within chunk (0..7)
  const int se = threadIdx.x & 31;   // element within B/C row pair (0..31)

  for (int l0 = 0; l0 < LSEQ; l0 += SCAN_CHUNK) {
    sBC[st][se] = x_dbl[(size_t)(l0 + st) * (DTRANK + 2 * NSTATE) + DTRANK + se];
    __syncthreads();

#pragma unroll
    for (int t = 0; t < SCAN_CHUNK; ++t) {
      const int l = l0 + t;
      const float dt = delta[(size_t)l * DIN + d];
      const float u  = xi_act[(size_t)l * DIN + d];
      const float du = dt * u;
      float yv = 0.0f;
#pragma unroll
      for (int n = 0; n < NSTATE; ++n) {
        h[n] = __expf(dt * Arow[n]) * h[n] + du * sBC[t][n];
        yv  += h[n] * sBC[t][NSTATE + n];
      }
      yv += u * Dd;
      const float rv = res[(size_t)l * (2 * DIN) + d];
      yv *= swish_f(rv);
      y_h[(size_t)l * DIN + d] = (_Float16)yv;
    }
    __syncthreads();   // protect sBC before next chunk's staging
  }
}

// ---------------------------------------------------------------------------
// Launcher
// ---------------------------------------------------------------------------
extern "C" void kernel_launch(void* const* d_in, const int* in_sizes, int n_in,
                              void* d_out, int out_size, void* d_ws, size_t ws_size,
                              hipStream_t stream) {
  (void)in_sizes; (void)n_in; (void)out_size; (void)ws_size;
  const float* x     = (const float*)d_in[0];   // L x DMODEL
  const float* W_in  = (const float*)d_in[1];   // DMODEL x 2*DIN
  const float* convw = (const float*)d_in[2];   // DIN x 1 x K
  const float* convb = (const float*)d_in[3];   // DIN
  const float* W_x   = (const float*)d_in[4];   // DIN x 96
  const float* W_dt  = (const float*)d_in[5];   // DTRANK x DIN
  const float* b_dt  = (const float*)d_in[6];   // DIN
  const float* A_log = (const float*)d_in[7];   // DIN x N
  const float* Dp    = (const float*)d_in[8];   // DIN
  const float* W_out = (const float*)d_in[9];   // DIN x DMODEL
  const float* b_out = (const float*)d_in[10];  // DMODEL
  float* out = (float*)d_out;                   // L x DMODEL

  // Workspace carve-out (256B aligned).
  char* ws = (char*)d_ws;
  auto carve = [&](size_t bytes) {
    char* p = ws;
    ws += (bytes + 255) & ~(size_t)255;
    return p;
  };
  const int NX = DTRANK + 2 * NSTATE;  // 96
  _Float16* xh     = (_Float16*)carve((size_t)LSEQ * DMODEL * 2);
  _Float16* WinT   = (_Float16*)carve((size_t)(2 * DIN) * DMODEL * 2);
  float*    C1     = (float*)   carve((size_t)LSEQ * (2 * DIN) * 4);
  float*    xi_act = (float*)   carve((size_t)LSEQ * DIN * 4);
  _Float16* xih    = (_Float16*)carve((size_t)LSEQ * DIN * 2);
  _Float16* WxT    = (_Float16*)carve((size_t)NX * DIN * 2);
  float*    xdbl   = (float*)   carve((size_t)LSEQ * NX * 4);
  _Float16* drh    = (_Float16*)carve((size_t)LSEQ * DTRANK * 2);
  _Float16* WdtT   = (_Float16*)carve((size_t)DIN * DTRANK * 2);
  float*    dlt    = (float*)   carve((size_t)LSEQ * DIN * 4);
  _Float16* yh     = (_Float16*)carve((size_t)LSEQ * DIN * 2);
  _Float16* WoutT  = (_Float16*)carve((size_t)DMODEL * DIN * 2);

  const int T = 256;
  auto blocks = [&](int n) { return (n + T - 1) / T; };

  // 0) f16 conversions / weight transposes
  cast_f32_f16_v4<<<blocks(LSEQ * DMODEL / 4), T, 0, stream>>>(x, xh, LSEQ * DMODEL / 4);
  transpose_cast_f16<<<blocks(DMODEL * 2 * DIN), T, 0, stream>>>(W_in, WinT, DMODEL, 2 * DIN);
  transpose_cast_f16<<<blocks(DIN * NX), T, 0, stream>>>(W_x, WxT, DIN, NX);
  transpose_cast_f16<<<blocks(DTRANK * DIN), T, 0, stream>>>(W_dt, WdtT, DTRANK, DIN);
  transpose_cast_f16<<<blocks(DIN * DMODEL), T, 0, stream>>>(W_out, WoutT, DIN, DMODEL);

  // 1) C1 = x @ W_in : (2048 x 1024) x (1024 x 4096)
  gemm_wmma_f16<2, 4, 0><<<dim3((2 * DIN) / 64, LSEQ / 32), 32, 0, stream>>>(
      xh, WinT, C1, nullptr, LSEQ, 2 * DIN, DMODEL);

  // 2) depthwise causal conv + swish on xi = C1[:, :DIN]
  conv_swish_kernel<<<blocks(LSEQ * DIN), T, 0, stream>>>(C1, convw, convb, xi_act, xih);

  // 3) x_dbl = xi @ W_x : (2048 x 2048) x (2048 x 96)
  gemm_wmma_f16<4, 2, 0><<<dim3(NX / 32, LSEQ / 64), 32, 0, stream>>>(
      xih, WxT, xdbl, nullptr, LSEQ, NX, DIN);

  // 4) delta_r (first 64 cols of x_dbl) -> f16
  slice_cast_f16_v4<<<blocks(LSEQ * DTRANK / 4), T, 0, stream>>>(xdbl, NX, drh, LSEQ, DTRANK);

  // 5) delta = softplus(delta_r @ W_dt + b_dt) : (2048 x 64) x (64 x 2048)
  gemm_wmma_f16<2, 4, 2><<<dim3(DIN / 64, LSEQ / 32), 32, 0, stream>>>(
      drh, WdtT, dlt, b_dt, LSEQ, DIN, DTRANK);

  // 6) selective scan (+ u*D, * swish(res)) -> y (f16)
  scan_kernel<<<DIN / T, T, 0, stream>>>(xi_act, dlt, xdbl, C1 + DIN, A_log, Dp, yh);

  // 7) out = y @ W_out + b_out : (2048 x 2048) x (2048 x 1024)
  gemm_wmma_f16<2, 4, 1><<<dim3(DMODEL / 64, LSEQ / 32), 32, 0, stream>>>(
      yh, WoutT, out, b_out, LSEQ, DMODEL, DIN);
}
